// GeometricProductConv1D_30442728194057
// MI455X (gfx1250) — matrix operand order repaired and checked by reading, not compiled, not run on hardware
//
#include <hip/hip_runtime.h>
#include <hip/hip_bf16.h>
#include <stdint.h>

// ---------------------------------------------------------------------------
// Geometric-product Conv1D (3D Euclidean GA, 8 blades) as implicit GEMM.
// M = 32*1024 = 32768, N = 64*8 = 512, K = 3*64*8 = 1536.
// Precision: split-bf16 (hi+lo) with f32 WMMA accumulation (~fp32 accuracy).
// Fast path: precomputed bf16 planes + async global->LDS staging (ASYNCcnt).
// ---------------------------------------------------------------------------

typedef __attribute__((ext_vector_type(16))) __bf16          v16bf;
typedef __attribute__((ext_vector_type(8)))  float           v8f;
typedef __attribute__((ext_vector_type(8)))  unsigned short  u16x8;

#define S_LEN   1024
#define S_PAD   (S_LEN + 2)  // one zero halo row on each side, per image
#define CI_     64
#define CO_     64
#define KW      3
#define KDIM    512          // CI*8  (k axis, per tap)
#define QDIM    512          // CO*8  (output channel axis)
#define NBATCH  32
#define MTOT    (NBATCH * S_LEN)

#define BM 128
#define BN 64
#define BK 32

// blade index <-> bitmask for blades ordered by (degree, mask):
// order = [0,1,2,4,3,5,6,7]; this table is its own inverse.
__device__ __constant__ int IDX_MASK[8] = {0, 1, 2, 4, 3, 5, 6, 7};

static __device__ inline unsigned short bf16_rn(float x) {
  union { float f; uint32_t u; } a; a.f = x;
  uint32_t u = a.u;
  uint32_t r = u + 0x7FFFu + ((u >> 16) & 1u);
  if ((u & 0x7F800000u) == 0x7F800000u) r = u;   // inf/nan: truncate
  return (unsigned short)(r >> 16);
}
static __device__ inline float bf16_f32(unsigned short h) {
  union { float f; uint32_t u; } a; a.u = ((uint32_t)h) << 16; return a.f;
}
static __device__ inline void split_bf16(float x, unsigned short& h, unsigned short& l) {
  h = bf16_rn(x);
  l = bf16_rn(x - bf16_f32(h));
}

union Frag { v16bf v; u16x8 u[2]; };

// ---------------------------------------------------------------------------
// Pass 1: fold Cayley table into the conv kernel, store transposed [t][q][k]
// as bf16 hi/lo planes. Each (bi,bo) pair maps to exactly one bk with sign +-1.
// ---------------------------------------------------------------------------
__global__ void keff_prep(const float* __restrict__ kern,          // [3,64,64,8]
                          unsigned short* __restrict__ keff_hi,    // [3,512,512]
                          unsigned short* __restrict__ keff_lo) {
  int gid = blockIdx.x * blockDim.x + threadIdx.x;
  if (gid >= KW * QDIM * KDIM) return;
  int k = gid & (KDIM - 1);
  int q = (gid >> 9) & (QDIM - 1);
  int t = gid >> 18;
  int ci = k >> 3, bi = k & 7;
  int co = q >> 3, bo = q & 7;
  int mi = IDX_MASK[bi];
  int mo = IDX_MASK[bo];
  int mk = mi ^ mo;              // unique bk mask contributing to bo
  int bk = IDX_MASK[mk];         // mask -> idx (same table)
  int swaps = 0;
  int xm = mi >> 1;
  while (xm) { swaps += __popc(xm & mk); xm >>= 1; }
  float sign = (swaps & 1) ? -1.0f : 1.0f;
  float v = sign * kern[(((t * CI_ + ci) * CO_ + co) << 3) + bk];
  unsigned short h, l;
  split_bf16(v, h, l);
  keff_hi[gid] = h;
  keff_lo[gid] = l;
}

// ---------------------------------------------------------------------------
// Pass 2 (fast path): split x into bf16 hi/lo planes with zero halo rows:
// layout [32][1026][512]; padded row 0 and 1025 of each image are zeros.
// ---------------------------------------------------------------------------
__global__ void xsplit(const float* __restrict__ x,          // [32][1024][512]
                       unsigned short* __restrict__ xh,      // [32][1026][512]
                       unsigned short* __restrict__ xl) {
  size_t gid = (size_t)blockIdx.x * blockDim.x + threadIdx.x;  // unit = 8 elems
  size_t base = gid * 8;
  if (base >= (size_t)NBATCH * S_PAD * KDIM) return;
  int    k    = (int)(base & (KDIM - 1));   // 8 | KDIM -> chunk stays in one row
  size_t rowp = base >> 9;                  // padded row = n*1026 + r
  int    r    = (int)(rowp % S_PAD);
  size_t n    = rowp / S_PAD;
  u16x8 h = {}, l = {};
  if (r >= 1 && r <= S_LEN) {
    const float* src = x + (((n * S_LEN) + (size_t)(r - 1)) << 9) + k;
    #pragma unroll
    for (int i = 0; i < 8; ++i) {
      unsigned short hh, ll;
      split_bf16(src[i], hh, ll);
      h[i] = hh; l[i] = ll;
    }
  }
  *(u16x8*)(xh + base) = h;
  *(u16x8*)(xl + base) = l;
}

// ---------------------------------------------------------------------------
// Pass 3 (fast path): implicit-GEMM conv. Block = 256 threads (8 wave32s),
// tile 128x64, wave grid 4(M) x 2(N), each wave 32x32 = 2x2 WMMA tiles.
// A tiles staged LDS via global_load_async_to_lds_b128 (pure DMA, no VALU).
// ---------------------------------------------------------------------------
__global__ __launch_bounds__(256)
void gp_conv_gemm_fast(const unsigned short* __restrict__ xh,  // [32][1026][512]
                       const unsigned short* __restrict__ xl,
                       const unsigned short* __restrict__ kh,  // [3][512][512] [t][q][k]
                       const unsigned short* __restrict__ kl,
                       const float* __restrict__ bias,         // [512]
                       float* __restrict__ out) {              // [32768, 512]
  __shared__ __attribute__((aligned(16))) unsigned short a_hi[BM][BK];
  __shared__ __attribute__((aligned(16))) unsigned short a_lo[BM][BK];

  const int tid   = threadIdx.x;
  const int lane  = tid & 31;
  const int wave  = tid >> 5;
  const int waveM = wave & 3;           // 0..3
  const int waveN = wave >> 2;          // 0..1
  const int half  = lane >> 4;          // 0 | 1
  const int lrow  = lane & 15;

  const int mTile = blockIdx.x * BM;    // 1024 % 128 == 0 -> no image straddle
  const int nImg  = mTile / S_LEN;
  const int sBase = mTile % S_LEN;
  const int nTile = blockIdx.y * BN;

  // low 32 bits of a generic LDS pointer are the in-wave LDS byte offset
  const unsigned ldsAh = (unsigned)(uintptr_t)&a_hi[0][0];
  const unsigned ldsAl = (unsigned)(uintptr_t)&a_lo[0][0];

  v8f acc[2][2] = {};

  for (int t = 0; t < KW; ++t) {
    // padded row index of tile row 0 for this tap: (sBase + 0) + t  (t-1 shift +1 halo)
    const size_t gRowBase = (((size_t)nImg * S_PAD) + (size_t)(sBase + t)) << 9;
    for (int kk = 0; kk < KDIM; kk += BK) {
      __syncthreads();   // previous tile fully consumed before overwrite
      // ---- stage A tile (128 rows x 64B per plane) with async DMA copies ----
      #pragma unroll
      for (int it = 0; it < 2; ++it) {
        int c   = tid + it * 256;        // chunk 0..511 (16B each)
        int row = c >> 2;
        int cb  = (c & 3) << 4;          // byte offset inside 64B tile row
        unsigned ldsOff = (unsigned)(row * (BK * 2) + cb);
        size_t gOff = gRowBase + (size_t)row * KDIM + (size_t)kk + (size_t)(cb >> 1);
        const unsigned short* gh = xh + gOff;
        const unsigned short* gl = xl + gOff;
        unsigned dh = ldsAh + ldsOff;
        unsigned dl = ldsAl + ldsOff;
        asm volatile("global_load_async_to_lds_b128 %0, %1, off"
                     :: "v"(dh), "v"(gh) : "memory");
        asm volatile("global_load_async_to_lds_b128 %0, %1, off"
                     :: "v"(dl), "v"(gl) : "memory");
      }
      asm volatile("s_wait_asynccnt 0" ::: "memory");
      __syncthreads();

      // ---- B fragments from global (3 MiB total, L2-resident) ----
      // ISA 16-bit B 32x16 layout: lane n<16 -> col n, K{0..7,16..23}; lane n+16 -> K{8..15,24..31}
      Frag bh[2], bl[2];
      const int kSel = kk + (half ? 8 : 0);
      #pragma unroll
      for (int sn = 0; sn < 2; ++sn) {
        int q = nTile + waveN * 32 + sn * 16 + lrow;
        const unsigned short* ph = kh + ((size_t)t * QDIM + q) * KDIM + kSel;
        const unsigned short* pl = kl + ((size_t)t * QDIM + q) * KDIM + kSel;
        bh[sn].u[0] = *(const u16x8*)(ph);
        bh[sn].u[1] = *(const u16x8*)(ph + 16);
        bl[sn].u[0] = *(const u16x8*)(pl);
        bl[sn].u[1] = *(const u16x8*)(pl + 16);
      }

      // ---- A fragments from LDS ----
      // ISA 16-bit A 16x32 layout: lane<16 -> K{0..7,16..23}; lane>=16 -> K{8..15,24..31}
      Frag ah[2], al[2];
      const int aBase = half ? 8 : 0;
      #pragma unroll
      for (int sm = 0; sm < 2; ++sm) {
        int m = waveM * 32 + sm * 16 + lrow;
        ah[sm].u[0] = *(const u16x8*)(&a_hi[m][aBase]);
        ah[sm].u[1] = *(const u16x8*)(&a_hi[m][aBase + 16]);
        al[sm].u[0] = *(const u16x8*)(&a_lo[m][aBase]);
        al[sm].u[1] = *(const u16x8*)(&a_lo[m][aBase + 16]);
      }

      // ---- 2x2 tiles x 3 split products = 12 WMMA per K-step ----
      #pragma unroll
      for (int sm = 0; sm < 2; ++sm)
        #pragma unroll
        for (int sn = 0; sn < 2; ++sn) {
          acc[sm][sn] = __builtin_amdgcn_wmma_f32_16x16x32_bf16(
              false, ah[sm].v, false, bh[sn].v, (short)0, acc[sm][sn], false, false);
          acc[sm][sn] = __builtin_amdgcn_wmma_f32_16x16x32_bf16(
              false, ah[sm].v, false, bl[sn].v, (short)0, acc[sm][sn], false, false);
          acc[sm][sn] = __builtin_amdgcn_wmma_f32_16x16x32_bf16(
              false, al[sm].v, false, bh[sn].v, (short)0, acc[sm][sn], false, false);
        }
    }
  }

  // ---- epilogue: bias add + NT streaming store ----
  #pragma unroll
  for (int sm = 0; sm < 2; ++sm) {
    int rowBase = mTile + waveM * 32 + sm * 16 + 8 * half;
    #pragma unroll
    for (int sn = 0; sn < 2; ++sn) {
      int q = nTile + waveN * 32 + sn * 16 + lrow;
      float b = bias[q];
      #pragma unroll
      for (int r = 0; r < 8; ++r) {
        __builtin_nontemporal_store(acc[sm][sn][r] + b,
                                    out + (size_t)(rowBase + r) * QDIM + q);
      }
    }
  }
}

// ---------------------------------------------------------------------------
// Fallback GEMM (ws too small for x planes): split x on the fly into LDS.
// ---------------------------------------------------------------------------
__global__ __launch_bounds__(256)
void gp_conv_gemm_inline(const float* __restrict__ x,                 // [32768, 512]
                         const unsigned short* __restrict__ keff_hi,  // [3,512,512]
                         const unsigned short* __restrict__ keff_lo,
                         const float* __restrict__ bias,
                         float* __restrict__ out) {
  __shared__ __attribute__((aligned(16))) unsigned short a_hi[BM][BK];
  __shared__ __attribute__((aligned(16))) unsigned short a_lo[BM][BK];

  const int tid   = threadIdx.x;
  const int lane  = tid & 31;
  const int wave  = tid >> 5;
  const int waveM = wave & 3;
  const int waveN = wave >> 2;
  const int half  = lane >> 4;
  const int lrow  = lane & 15;

  const int mTile = blockIdx.x * BM;
  const int nImg  = mTile / S_LEN;
  const int sBase = mTile % S_LEN;
  const int nTile = blockIdx.y * BN;

  v8f acc[2][2] = {};

  for (int t = 0; t < KW; ++t) {
    const int sOff = t - 1;
    for (int kk = 0; kk < KDIM; kk += BK) {
      __syncthreads();
      #pragma unroll
      for (int it = 0; it < 4; ++it) {
        int e4  = tid + it * 256;
        int row = e4 >> 3;
        int c4  = (e4 & 7) << 2;
        int srow = sBase + row + sOff;
        float4 v = make_float4(0.f, 0.f, 0.f, 0.f);
        if ((unsigned)srow < (unsigned)S_LEN) {
          v = *(const float4*)(x + (size_t)(nImg * S_LEN + srow) * KDIM + kk + c4);
        }
        unsigned short h, l;
        split_bf16(v.x, h, l); a_hi[row][c4 + 0] = h; a_lo[row][c4 + 0] = l;
        split_bf16(v.y, h, l); a_hi[row][c4 + 1] = h; a_lo[row][c4 + 1] = l;
        split_bf16(v.z, h, l); a_hi[row][c4 + 2] = h; a_lo[row][c4 + 2] = l;
        split_bf16(v.w, h, l); a_hi[row][c4 + 3] = h; a_lo[row][c4 + 3] = l;
      }
      __syncthreads();

      Frag bh[2], bl[2];
      const int kSel = kk + (half ? 8 : 0);
      #pragma unroll
      for (int sn = 0; sn < 2; ++sn) {
        int q = nTile + waveN * 32 + sn * 16 + lrow;
        const unsigned short* ph = keff_hi + ((size_t)t * QDIM + q) * KDIM + kSel;
        const unsigned short* pl = keff_lo + ((size_t)t * QDIM + q) * KDIM + kSel;
        bh[sn].u[0] = *(const u16x8*)(ph);
        bh[sn].u[1] = *(const u16x8*)(ph + 16);
        bl[sn].u[0] = *(const u16x8*)(pl);
        bl[sn].u[1] = *(const u16x8*)(pl + 16);
      }

      Frag ah[2], al[2];
      const int aBase = half ? 8 : 0;
      #pragma unroll
      for (int sm = 0; sm < 2; ++sm) {
        int m = waveM * 32 + sm * 16 + lrow;
        ah[sm].u[0] = *(const u16x8*)(&a_hi[m][aBase]);
        ah[sm].u[1] = *(const u16x8*)(&a_hi[m][aBase + 16]);
        al[sm].u[0] = *(const u16x8*)(&a_lo[m][aBase]);
        al[sm].u[1] = *(const u16x8*)(&a_lo[m][aBase + 16]);
      }

      #pragma unroll
      for (int sm = 0; sm < 2; ++sm)
        #pragma unroll
        for (int sn = 0; sn < 2; ++sn) {
          acc[sm][sn] = __builtin_amdgcn_wmma_f32_16x16x32_bf16(
              false, ah[sm].v, false, bh[sn].v, (short)0, acc[sm][sn], false, false);
          acc[sm][sn] = __builtin_amdgcn_wmma_f32_16x16x32_bf16(
              false, ah[sm].v, false, bl[sn].v, (short)0, acc[sm][sn], false, false);
          acc[sm][sn] = __builtin_amdgcn_wmma_f32_16x16x32_bf16(
              false, al[sm].v, false, bh[sn].v, (short)0, acc[sm][sn], false, false);
        }
    }
  }

  #pragma unroll
  for (int sm = 0; sm < 2; ++sm) {
    int rowBase = mTile + waveM * 32 + sm * 16 + 8 * half;
    #pragma unroll
    for (int sn = 0; sn < 2; ++sn) {
      int q = nTile + waveN * 32 + sn * 16 + lrow;
      float b = bias[q];
      #pragma unroll
      for (int r = 0; r < 8; ++r) {
        __builtin_nontemporal_store(acc[sm][sn][r] + b,
                                    out + (size_t)(rowBase + r) * QDIM + q);
      }
    }
  }
}

// ---------------------------------------------------------------------------
extern "C" void kernel_launch(void* const* d_in, const int* in_sizes, int n_in,
                              void* d_out, int out_size, void* d_ws, size_t ws_size,
                              hipStream_t stream) {
  const float* x    = (const float*)d_in[0];   // [32,1024,64,8]
  const float* kern = (const float*)d_in[1];   // [3,64,64,8]
  const float* bias = (const float*)d_in[2];   // [64,8]
  float* out = (float*)d_out;                  // [32,1024,64,8]

  const size_t keffElems = (size_t)KW * QDIM * KDIM;        // 786432
  const size_t xpadElems = (size_t)NBATCH * S_PAD * KDIM;   // 16,809,984
  const size_t wsFast    = (keffElems * 2 + xpadElems * 2) * sizeof(unsigned short);

  unsigned short* keff_hi = (unsigned short*)d_ws;
  unsigned short* keff_lo = keff_hi + keffElems;

  keff_prep<<<(int)((keffElems + 255) / 256), 256, 0, stream>>>(kern, keff_hi, keff_lo);

  dim3 grid(MTOT / BM, QDIM / BN);             // 256 x 8 = 2048 blocks

  if (ws_size >= wsFast) {
    unsigned short* x_hi = keff_lo + keffElems;
    unsigned short* x_lo = x_hi + xpadElems;
    const size_t chunks = xpadElems / 8;       // 2,101,248
    xsplit<<<(int)((chunks + 255) / 256), 256, 0, stream>>>(x, x_hi, x_lo);
    gp_conv_gemm_fast<<<grid, 256, 0, stream>>>(x_hi, x_lo, keff_hi, keff_lo, bias, out);
  } else {
    gp_conv_gemm_inline<<<grid, 256, 0, stream>>>(x, keff_hi, keff_lo, bias, out);
  }
}
